// LTCCell_300647710891
// MI455X (gfx1250) — compile-verified, matrix operand order checked by reading, not run on hardware
//
#include <hip/hip_runtime.h>

typedef __attribute__((ext_vector_type(16))) _Float16 v16h;
typedef __attribute__((ext_vector_type(8)))  float    v8f;
typedef __attribute__((ext_vector_type(4)))  _Float16 h4;   // 8-byte packed params

#define N_      256
#define S_      256
#define B_      512
#define UNFOLDS 6
#define EPS_    1e-8f
#define L2E_    1.4426950408889634f

#if __has_builtin(__builtin_amdgcn_exp2f)
#define EXP2F(x) __builtin_amdgcn_exp2f(x)
#else
#define EXP2F(x) exp2f(x)
#endif
#if __has_builtin(__builtin_amdgcn_rcpf)
#define RCPF(x) __builtin_amdgcn_rcpf(x)
#else
#define RCPF(x) (1.0f/(x))
#endif

// Pack {sigma*log2e, sigma*mu*log2e, w, w*erev} as 4 x f16 (8 bytes/element).
// Inner loop becomes:
//   t = (f32)p.y - (f32)p.x * v   (v_fma_mix_f32)
//   e = 2^t; r = rcp(1+e)         (v_exp_f32, v_rcp_f32)
//   act = p.z * (f16)r; nact = p.w * (f16)r  (f16 muls feeding WMMA B operand)
__global__ __launch_bounds__(256) void pack_params_kernel(
    const float* __restrict__ sig, const float* __restrict__ mu,
    const float* __restrict__ w,   const float* __restrict__ erev,
    h4* __restrict__ pack, int n)
{
    int idx = blockIdx.x * blockDim.x + threadIdx.x;
    if (idx < n) {
        float s = sig[idx], m = mu[idx], ww = w[idx], e = erev[idx];
        h4 o;
        o.x = (_Float16)(s * L2E_);
        o.y = (_Float16)(s * m * L2E_);
        o.z = (_Float16)ww;
        o.w = (_Float16)(ww * e);
        pack[idx] = o;
    }
}

// Sensory reduction: one block per batch element, thread = output neuron j,
// loop over s with the (b)-row of inputs staged in LDS (broadcast reads).
__global__ __launch_bounds__(256) void sensory_kernel(
    const float* __restrict__ input, const float* __restrict__ iw,
    const float* __restrict__ ib,    const h4* __restrict__ packS,
    float* __restrict__ wnum, float* __restrict__ wden)
{
    __shared__ float xin[S_];
    int j = threadIdx.x;
    int b = blockIdx.x;
    xin[j] = input[b * S_ + j] * iw[j] + ib[j];
    __syncthreads();

    float num = 0.f, den = 0.f;
    for (int s = 0; s < S_; ++s) {
        h4    p = packS[s * N_ + j];          // global_load_b64
        float x = xin[s];
        float t = (float)p.y - (float)p.x * x;
        float e = EXP2F(t);
        float r = RCPF(1.0f + e);
        den += (float)p.z * r;
        num += (float)p.w * r;
    }
    wnum[b * N_ + j] = num;
    wden[b * N_ + j] = den;
}

// ODE unfolds. Block = 8 waves handling 4 batch elements (2 waves per b,
// each wave owns 8 of the 16 j-tiles). Reduction over presynaptic i is done
// with v_wmma_f32_16x16x32_f16: A = ones(16x32), B = activations (32 i x 16 j),
// so D[m,j] = sum_i act[i,j] for every m; f32 accumulator carried across the
// 8 K-steps covering i = 0..255.
__global__ __launch_bounds__(256) void ode_kernel(
    const float* __restrict__ hx,    const float* __restrict__ ts,
    const float* __restrict__ gleak, const float* __restrict__ vleak,
    const float* __restrict__ cm,    const h4* __restrict__ packN,
    const float* __restrict__ wnum_s, const float* __restrict__ wden_s,
    float* __restrict__ out)
{
    __shared__ __align__(16) float vbuf[2][4][N_];   // double-buffered v
    __shared__ float gl[N_], vl[N_], cmv[N_];
    __shared__ float wns[4][N_], wds[4][N_];

    const int tid   = threadIdx.x;
    const int wave  = tid >> 5;
    const int ln    = tid & 31;
    const int b_loc = wave >> 1;      // 4 batches per block
    const int jhalf = wave & 1;       // 2 waves split the 16 j-tiles
    const int b0    = blockIdx.x * 4;
    const int b     = b0 + b_loc;

    gl[tid]  = gleak[tid];
    vl[tid]  = vleak[tid];
    cmv[tid] = cm[tid];
#pragma unroll
    for (int r = 0; r < 4; ++r) {
        vbuf[0][r][tid] = hx[(b0 + r) * N_ + tid];
        wns[r][tid]     = wnum_s[(b0 + r) * N_ + tid];
        wds[r][tid]     = wden_s[(b0 + r) * N_ + tid];
    }
    __syncthreads();

    const float cmt_scale = (float)UNFOLDS / (ts[b] + 1.0f);

    v16h ones;
#pragma unroll
    for (int t = 0; t < 16; ++t) ones[t] = (_Float16)1.0f;

    const int half = ln >> 4;    // which 16 of the 32-wide K block this lane covers
    const int jn   = ln & 15;    // column within the j-tile

    int cb = 0;
    for (int u = 0; u < UNFOLDS; ++u) {
        const float* vcur = &vbuf[cb][b_loc][0];
        for (int jt = jhalf * 8; jt < jhalf * 8 + 8; ++jt) {
            const int j = jt * 16 + jn;
            v8f cd = {};   // denominator accumulator (16x16 f32, all rows equal)
            v8f cn = {};   // numerator accumulator
            for (int ks = 0; ks < 8; ++ks) {
                const int ibase = ks * 32 + half * 16;
                // broadcast-load 16 presynaptic voltages (ds_load_b128 x4)
                float vi[16];
                const float4* v4p = reinterpret_cast<const float4*>(vcur + ibase);
#pragma unroll
                for (int q = 0; q < 4; ++q) {
                    float4 t4 = v4p[q];
                    vi[4 * q + 0] = t4.x; vi[4 * q + 1] = t4.y;
                    vi[4 * q + 2] = t4.z; vi[4 * q + 3] = t4.w;
                }
                v16h bd, bn;
#pragma unroll
                for (int t = 0; t < 16; ++t) {
                    h4    p  = packN[(ibase + t) * N_ + j];   // global_load_b64
                    float tt = (float)p.y - (float)p.x * vi[t]; // v_fma_mix_f32
                    float e  = EXP2F(tt);                     // v_exp_f32
                    float r  = RCPF(1.0f + e);                // v_rcp_f32
                    _Float16 rh = (_Float16)r;
                    bd[t] = p.z * rh;                         // act        (f16)
                    bn[t] = p.w * rh;                         // act * erev (f16)
                }
                cd = __builtin_amdgcn_wmma_f32_16x16x32_f16(
                        false, ones, false, bd, (short)0, cd, false, false);
                cn = __builtin_amdgcn_wmma_f32_16x16x32_f16(
                        false, ones, false, bn, (short)0, cn, false, false);
            }
            if (ln < 16) {
                float den  = cd[0] + wds[b_loc][j];
                float num  = cn[0] + wns[b_loc][j];
                float cmt  = cmv[j] * cmt_scale;
                float g    = gl[j];
                float vo   = vcur[j];
                float vnew = (cmt * vo + g * vl[j] + num) * RCPF(cmt + g + den + EPS_);
                vbuf[cb ^ 1][b_loc][j] = vnew;
            }
        }
        __syncthreads();
        cb ^= 1;
    }

#pragma unroll
    for (int r = 0; r < 4; ++r)
        out[(b0 + r) * N_ + tid] = vbuf[cb][r][tid];
}

extern "C" void kernel_launch(void* const* d_in, const int* in_sizes, int n_in,
                              void* d_out, int out_size, void* d_ws, size_t ws_size,
                              hipStream_t stream) {
    (void)in_sizes; (void)n_in; (void)out_size; (void)ws_size;

    const float* input = (const float*)d_in[0];
    const float* hx    = (const float*)d_in[1];
    const float* ts    = (const float*)d_in[2];
    const float* gleak = (const float*)d_in[3];
    const float* vleak = (const float*)d_in[4];
    const float* cm    = (const float*)d_in[5];
    const float* sigma = (const float*)d_in[6];
    const float* mu    = (const float*)d_in[7];
    const float* w     = (const float*)d_in[8];
    const float* erev  = (const float*)d_in[9];
    const float* ssig  = (const float*)d_in[10];
    const float* smu   = (const float*)d_in[11];
    const float* sw    = (const float*)d_in[12];
    const float* serev = (const float*)d_in[13];
    const float* iw    = (const float*)d_in[14];
    const float* ib    = (const float*)d_in[15];
    float* out = (float*)d_out;

    char* ws    = (char*)d_ws;
    h4*   packN = (h4*)(ws);                                 // 512 KB
    h4*   packS = (h4*)(ws + (512 << 10));                   // 512 KB
    float* wnum = (float*)(ws + (1 << 20));                  // 512 KB
    float* wden = (float*)(ws + (1 << 20) + (512 << 10));    // 512 KB

    pack_params_kernel<<<(N_ * N_) / 256, 256, 0, stream>>>(sigma, mu, w, erev, packN, N_ * N_);
    pack_params_kernel<<<(S_ * N_) / 256, 256, 0, stream>>>(ssig, smu, sw, serev, packS, S_ * N_);
    sensory_kernel<<<B_, 256, 0, stream>>>(input, iw, ib, packS, wnum, wden);
    ode_kernel<<<B_ / 4, 256, 0, stream>>>(hx, ts, gleak, vleak, cm, packN, wnum, wden, out);
}
